// MVSNet_8787503087847
// MI455X (gfx1250) — matrix operand order repaired and checked by reading, not compiled, not run on hardware
//
#include <hip/hip_runtime.h>

// ---------------------------------------------------------------------------
// MVSNet pipeline for gfx1250 (MI455X).
// All conv2d / conv3d / deconv3d layers run through ONE implicit-GEMM kernel
// using v_wmma_f32_16x16x32_f16 (f16 in, f32 accumulate). Activations stay in
// f32 in HBM; conversion to f16 happens during the cooperative LDS stage.
// v3: template-split conv/deconv, fully branchless predicated im2col loads
//     (clamped address + cndmask, no exec-mask diamonds), one-add addressing
//     via hoisted flat base offsets. Double-buffered tiles, 1 barrier/K-step.
// ---------------------------------------------------------------------------

typedef __attribute__((ext_vector_type(8)))  _Float16 v8h;
typedef __attribute__((ext_vector_type(16))) _Float16 v16h;
typedef __attribute__((ext_vector_type(8)))  float    v8f;

#define LRELU_F  0.01f
#define BN_EPS_F 1e-5f

#define HF   64
#define WF   80
#define HW2D (HF * WF)       // 5120
#define DPL  128
#define NVOL (DPL * HW2D)    // 655360
#define HWIMG (256 * 320)    // 81920

#define KP_MAX 1792          // max padded K across all layers (64*27=1728)

static __host__ __device__ inline int ceil_div(int a, int b) { return (a + b - 1) / b; }

// ---------------------------------------------------------------------------
// Pack conv / deconv weights into an f16 matrix A[M16][Kp], zero padded.
// conv:   W[M][Cin][kd][kh][kw]                -> A[m][((ci*kd+a)*kh+b)*kw+c]
// deconv: W[Cin][M][kd][kh][kw] (flip+transp)  -> A[m][...] = W[ci][m][kd-1-a][kh-1-b][kw-1-c]
// ---------------------------------------------------------------------------
__global__ void k_pack_weights(const float* __restrict__ W, _Float16* __restrict__ A,
                               int M, int K, int Kp, int M16,
                               int Cin, int kd, int kh, int kw, int deconv) {
    int idx = blockIdx.x * blockDim.x + threadIdx.x;
    if (idx >= M16 * Kp) return;
    int m = idx / Kp, k = idx % Kp;
    float v = 0.0f;
    if (m < M && k < K) {
        int khw = kh * kw, kvol = kd * khw;
        int ci = k / kvol; int rem = k % kvol;
        int a = rem / khw; rem %= khw;
        int b = rem / kw;  int c = rem % kw;
        if (!deconv) {
            v = W[(((size_t)m * Cin + ci) * kd + a) * (size_t)khw + (size_t)b * kw + c];
        } else {
            v = W[(((size_t)ci * M + m) * kd + (kd - 1 - a)) * (size_t)khw +
                  (size_t)(kh - 1 - b) * kw + (kw - 1 - c)];
        }
    }
    A[idx] = (_Float16)v;
}

// ---------------------------------------------------------------------------
// Implicit-GEMM conv kernel. Block = 128 threads (4 waves).
// Block tile: 16 (M) x 64 (N). Each wave -> one 16x16 WMMA tile.
// X layout: [Cin][inD][inH][inW]  (2D convs fold the view-batch into inD, kd=1)
// Y layout: [M][outD*outH*outW]
// DECONV: lhs_dilation=2, pad_lo=1 semantics (output = 2 * input dims).
// ---------------------------------------------------------------------------
template <int DECONV>
__global__ __launch_bounds__(128)
void k_conv_gemm_wmma(const float* __restrict__ X,
                      const _Float16* __restrict__ Aw,   // [M16][Kp] packed f16
                      const float* __restrict__ bias,    // nullptr or [M]
                      float* __restrict__ Y,
                      int M, int Kp, int K,
                      int Cin, int inD, int inH, int inW,
                      int outD, int outH, int outW,
                      int kd, int kh, int kw,
                      int sd, int sh, int sw,
                      int pd, int ph, int pw) {
    __shared__ __align__(16) _Float16 As[2][16 * 32];   // A tile, row-major [m][k]
    __shared__ __align__(16) _Float16 Bs[2][64 * 32];   // B tile, n-major  [n][k]
    __shared__ int kOff[KP_MAX];                        // k -> ci*inD*inH*inW
    __shared__ int kABC[KP_MAX];                        // k -> (a<<16)|(b<<8)|c

    const int t    = threadIdx.x;
    const int wave = t >> 5;
    const int lane = t & 31;
    const int lh   = lane >> 4;   // half-wave selector
    const int lm   = lane & 15;

    const int n0 = blockIdx.x * 64;
    const int m0 = blockIdx.y * 16;

    const int outHW = outH * outW;
    const int inHW  = inH * inW;
    const int N     = outD * outHW;
    const int khw   = kh * kw;
    const int kvol  = kd * khw;

    // ---- build K-decode table once (divisions leave the hot loop) ----------
    for (int k = t; k < Kp; k += 128) {
        int off = 0, abc;
        if (k < K) {
            int ci = k / kvol; int rem = k % kvol;
            int a = rem / khw; rem %= khw;
            int b = rem / kw;  int c = rem % kw;
            off = ci * inD * inHW;
            abc = (a << 16) | (b << 8) | c;
        } else {
            abc = (255 << 16) | (255 << 8) | 255;   // pad: always out of bounds
        }
        kOff[k] = off; kABC[k] = abc;
    }

    // ---- hoist n -> (z,y,x) decode out of the K loop.
    // pk[j]: packed biased base coords (for bounds tests only).
    // pb[j]: flat base offset izb*inHW + iyb*inW + ixb (conv address = 1 add).
    // Sentinel (n>=N): huge coords -> every bounds check fails.
    const int kk = t & 31;        // B-stage k row
    const int ng = t >> 5;        // B-stage n group (16 cols each)
    int pk[16], pb[16];
#pragma unroll
    for (int j = 0; j < 16; ++j) {
        int n = n0 + ng * 16 + j;
        int v = (1023 << 20) | (1023 << 10) | 1023;
        int fb = 0;
        if (n < N) {
            int z = n / outHW; int r2 = n % outHW;
            int y = r2 / outW; int x = r2 % outW;
            int izb, iyb, ixb;
            if (!DECONV) { izb = z * sd - pd; iyb = y * sh - ph; ixb = x * sw - pw; }
            else         { izb = z - 1;       iyb = y - 1;       ixb = x - 1; }
            v  = ((izb + 4) << 20) | ((iyb + 4) << 10) | (ixb + 4);
            fb = izb * inHW + iyb * inW + ixb;
        }
        pk[j] = v; pb[j] = fb;
    }
    __syncthreads();   // decode table ready

    // ---- tile stagers (double buffered) ------------------------------------
    auto stageA = [&](int k0, int buf) {
        int e = t * 4;            // 4 halfs per thread, 512 total
        int r = e >> 5;
        int c = e & 31;
        float2 v = *(const float2*)(Aw + (size_t)(m0 + r) * Kp + (k0 + c));
        *(float2*)(&As[buf][r * 32 + c]) = v;
    };
    auto stageB = [&](int k0, int buf) {
        int off = kOff[k0 + kk];
        int abc = kABC[k0 + kk];
        int a = abc >> 16, b = (abc >> 8) & 255, c = abc & 255;
        int doff = off + a * inHW + b * inW + c;   // conv: per-K flat offset
        _Float16* Bb = &Bs[buf][0];
#pragma unroll
        for (int j = 0; j < 16; ++j) {
            int pkj = pk[j];
            int izb = (pkj >> 20) - 4;
            int iyb = ((pkj >> 10) & 1023) - 4;
            int ixb = (pkj & 1023) - 4;
            bool ok; int addr;
            if (!DECONV) {
                int iz = izb + a, iy = iyb + b, ix = ixb + c;
                ok = ((unsigned)iz < (unsigned)inD) &
                     ((unsigned)iy < (unsigned)inH) &
                     ((unsigned)ix < (unsigned)inW);
                addr = doff + pb[j];               // == off + iz*inHW + iy*inW + ix
            } else {
                int tz = izb + a, ty = iyb + b, tx = ixb + c;
                int iz = tz >> 1, iy = ty >> 1, ix = tx >> 1;
                ok = (tz >= 0) & (ty >= 0) & (tx >= 0) &
                     (((tz | ty | tx) & 1) == 0) &
                     (iz < inD) & (iy < inH) & (ix < inW);
                addr = off + iz * inHW + iy * inW + ix;
            }
            addr = ok ? addr : 0;                  // branchless predication
            float xv = X[addr];                    // unconditional load
            _Float16 val = ok ? (_Float16)xv : (_Float16)0.0f;
            Bb[(ng * 16 + j) * 32 + kk] = val;
        }
    };

    v8f acc = {};

    stageA(0, 0);
    stageB(0, 0);
    __syncthreads();

    for (int k0 = 0; k0 < Kp; k0 += 32) {
        int cb = (k0 >> 5) & 1;
        if (k0 + 32 < Kp) {       // prefetch next tile into the other buffer
            stageA(k0 + 32, cb ^ 1);
            stageB(k0 + 32, cb ^ 1);
        }
        // ---- fragment gather per ISA wave32 layout, 2x b128 LDS reads each
        // A: lanes 0-15 -> K [0..7]+[16..23]; lanes 16-31 shifted by 8.
        const _Float16* ap = &As[cb][0] + lm * 32 + 8 * lh;
        v8h a0 = *(const v8h*)(ap);
        v8h a1 = *(const v8h*)(ap + 16);
        v16h afrag = __builtin_shufflevector(a0, a1, 0, 1, 2, 3, 4, 5, 6, 7,
                                             8, 9, 10, 11, 12, 13, 14, 15);
        // B: n = wave*16 + (lane&15); K = [0..7]+[16..23] for lanes 0-15, shifted by 8 for 16-31
        const _Float16* bp = &Bs[cb][0] + (wave * 16 + lm) * 32 + 8 * lh;
        v8h b0 = *(const v8h*)(bp);
        v8h b1 = *(const v8h*)(bp + 16);
        v16h bfrag = __builtin_shufflevector(b0, b1, 0, 1, 2, 3, 4, 5, 6, 7,
                                             8, 9, 10, 11, 12, 13, 14, 15);

        acc = __builtin_amdgcn_wmma_f32_16x16x32_f16(
            false, afrag, false, bfrag, (short)0, acc, false, false);
        __syncthreads();
    }

    // ---- epilogue: C layout VGPR r -> m = r + 8*lh, n = lane&15 (+wave*16)
    int n = n0 + wave * 16 + lm;
    if (n < N) {
        for (int r = 0; r < 8; ++r) {
            int m = m0 + r + 8 * lh;
            if (m < M) {
                float v = acc[r];
                if (bias) v += bias[m];
                Y[(size_t)m * N + n] = v;
            }
        }
    }
}

// ---------------------------------------------------------------------------
// BatchNorm statistics: per-channel sum / sumsq via block reduce + atomics.
// X layout [C][N]; grid.y = channel.
// ---------------------------------------------------------------------------
__global__ void k_bn_stats(const float* __restrict__ X, float* __restrict__ sum,
                           float* __restrict__ sumsq, int N) {
    int c = blockIdx.y;
    const float* xc = X + (size_t)c * N;
    float s = 0.f, q = 0.f;
    for (int i = blockIdx.x * blockDim.x + threadIdx.x; i < N;
         i += blockDim.x * gridDim.x) {
        float v = xc[i]; s += v; q += v * v;
    }
    __shared__ float sh[256], shq[256];
    sh[threadIdx.x] = s; shq[threadIdx.x] = q;
    __syncthreads();
    for (int off = 128; off > 0; off >>= 1) {
        if (threadIdx.x < off) {
            sh[threadIdx.x]  += sh[threadIdx.x + off];
            shq[threadIdx.x] += shq[threadIdx.x + off];
        }
        __syncthreads();
    }
    if (threadIdx.x == 0) {
        atomicAdd(&sum[c], sh[0]);
        atomicAdd(&sumsq[c], shq[0]);
    }
}

// ---------------------------------------------------------------------------
// BN scale/shift + leaky ReLU + optional skip add.  Y = skip + lrelu(bn(X))
// ---------------------------------------------------------------------------
__global__ void k_bn_apply(const float* __restrict__ X, const float* __restrict__ sum,
                           const float* __restrict__ sumsq, const float* __restrict__ g,
                           const float* __restrict__ beta, const float* __restrict__ skip,
                           float* __restrict__ Y, int N, float invN) {
    int c = blockIdx.y;
    float m   = sum[c] * invN;
    float var = sumsq[c] * invN - m * m;
    float sc  = g[c] * rsqrtf(var + BN_EPS_F);
    float shf = beta[c] - m * sc;
    const float* xc = X + (size_t)c * N;
    float* yc = Y + (size_t)c * N;
    const float* sk = skip ? skip + (size_t)c * N : nullptr;
    for (int i = blockIdx.x * blockDim.x + threadIdx.x; i < N;
         i += blockDim.x * gridDim.x) {
        float y = xc[i] * sc + shf;
        y = (y >= 0.f) ? y : LRELU_F * y;
        if (sk) y += sk[i];
        yc[i] = y;
    }
}

// ---------------------------------------------------------------------------
// Transpose imgs [V][C][HW] (NCHW, V=3,C=3) -> X0 [C][V][HW] so the feature
// net can fold the view-batch into the depth axis (kd=1 conv3d).
// ---------------------------------------------------------------------------
__global__ void k_transpose_imgs(const float* __restrict__ in, float* __restrict__ out) {
    int idx = blockIdx.x * blockDim.x + threadIdx.x;
    if (idx >= 9 * HWIMG) return;
    int vc = idx / HWIMG, p = idx % HWIMG;
    int v = vc / 3, c = vc % 3;
    out[((size_t)c * 3 + v) * HWIMG + p] = in[idx];
}

// Bilinear downsample 256x320 -> 64x80 (half-pixel centers), layout [v*3+c][HW2D]
__global__ void k_resize_imgs(const float* __restrict__ imgs, float* __restrict__ out) {
    int idx = blockIdx.x * blockDim.x + threadIdx.x;
    if (idx >= 9 * HW2D) return;
    int vc = idx / HW2D, p = idx % HW2D;
    int y = p / WF, x = p % WF;
    float fy = (y + 0.5f) * 4.0f - 0.5f;
    float fx = (x + 0.5f) * 4.0f - 0.5f;
    int y0 = (int)floorf(fy), x0 = (int)floorf(fx);
    float wy = fy - y0, wx = fx - x0;
    int y1 = y0 + 1, x1 = x0 + 1;
    y0 = min(max(y0, 0), 255); y1 = min(max(y1, 0), 255);
    x0 = min(max(x0, 0), 319); x1 = min(max(x1, 0), 319);
    const float* s = imgs + (size_t)vc * HWIMG;
    float v = s[y0 * 320 + x0] * (1 - wx) * (1 - wy) + s[y0 * 320 + x1] * wx * (1 - wy) +
              s[y1 * 320 + x0] * (1 - wx) * wy       + s[y1 * 320 + x1] * wx * wy;
    out[idx] = v;
}

// depth_values: near + (far-near)*linspace(0,1,128); write ws + d_out tail
__global__ void k_depth_vals(const float* __restrict__ nf, float* __restrict__ dv,
                             float* __restrict__ dout) {
    int d = threadIdx.x;
    if (d < DPL) {
        float v = nf[0] + (nf[1] - nf[0]) * ((float)d / (float)(DPL - 1));
        dv[d] = v; dout[d] = v;
    }
}

// Init cost volume from reference view: sum=feat0, sq=feat0^2, img chans, count=1
__global__ void k_init_vol(const float* __restrict__ feats,   // [32][3][HW2D]
                           const float* __restrict__ imgr,    // [9][HW2D]
                           float* __restrict__ vol41, float* __restrict__ volsq,
                           float* __restrict__ count) {
    int idx = blockIdx.x * blockDim.x + threadIdx.x;
    if (idx >= NVOL) return;
    int p = idx % HW2D;
    for (int c = 0; c < 32; ++c) {
        float s = feats[((size_t)c * 3 + 0) * HW2D + p];
        vol41[(size_t)(9 + c) * NVOL + idx] = s;
        volsq[(size_t)c * NVOL + idx] = s * s;
    }
    for (int c = 0; c < 3; ++c)
        vol41[(size_t)c * NVOL + idx] = imgr[(size_t)c * HW2D + p];
    count[idx] = 1.0f;
}

// Homography warp + bilinear grid-sample for one source view v (1 or 2).
__global__ void k_warp_view(const float* __restrict__ feats,  // [32][3][HW2D]
                            const float* __restrict__ imgr,   // [9][HW2D]
                            const float* __restrict__ proj,   // [3][3][4]
                            const float* __restrict__ dv,
                            float* __restrict__ vol41, float* __restrict__ volsq,
                            float* __restrict__ count, int v) {
    int idx = blockIdx.x * blockDim.x + threadIdx.x;
    if (idx >= NVOL) return;
    int d = idx / HW2D, p = idx % HW2D;
    int y = p / WF, x = p % WF;
    const float* P = proj + v * 12;
    float fx = (float)x, fy = (float)y;
    float bx = P[0] * fx + P[1] * fy + P[2];
    float by = P[4] * fx + P[5] * fy + P[6];
    float bz = P[8] * fx + P[9] * fy + P[10];
    float invd = 1.0f / dv[d];
    float sx = bx + P[3] * invd, sy = by + P[7] * invd, sz = bz + P[11] * invd;
    float px = sx / sz, py = sy / sz;
    float gx = px / ((WF - 1) * 0.5f) - 1.0f;
    float gy = py / ((HF - 1) * 0.5f) - 1.0f;
    bool inm = (gx > -1.f) & (gx < 1.f) & (gy > -1.f) & (gy < 1.f);
    float ux = (gx + 1.f) * 0.5f * (WF - 1);
    float uy = (gy + 1.f) * 0.5f * (HF - 1);
    float x0f = floorf(ux), y0f = floorf(uy);
    float wx = ux - x0f, wy = uy - y0f;
    int x0 = (int)x0f, y0 = (int)y0f, x1 = x0 + 1, y1 = y0 + 1;
    bool vx0 = (x0 >= 0) & (x0 <= WF - 1), vx1 = (x1 >= 0) & (x1 <= WF - 1);
    bool vy0 = (y0 >= 0) & (y0 <= HF - 1), vy1 = (y1 >= 0) & (y1 <= HF - 1);
    int cx0 = min(max(x0, 0), WF - 1), cx1 = min(max(x1, 0), WF - 1);
    int cy0 = min(max(y0, 0), HF - 1), cy1 = min(max(y1, 0), HF - 1);
    float m00 = (vx0 & vy0) ? (1 - wx) * (1 - wy) : 0.f;
    float m01 = (vx0 & vy1) ? (1 - wx) * wy : 0.f;
    float m10 = (vx1 & vy0) ? wx * (1 - wy) : 0.f;
    float m11 = (vx1 & vy1) ? wx * wy : 0.f;
    int o00 = cy0 * WF + cx0, o01 = cy1 * WF + cx0;
    int o10 = cy0 * WF + cx1, o11 = cy1 * WF + cx1;
    for (int c = 0; c < 32; ++c) {
        const float* f = feats + ((size_t)c * 3 + v) * HW2D;
        float s = f[o00] * m00 + f[o01] * m01 + f[o10] * m10 + f[o11] * m11;
        vol41[(size_t)(9 + c) * NVOL + idx] += s;
        volsq[(size_t)c * NVOL + idx] += s * s;
    }
    for (int c = 0; c < 3; ++c) {
        const float* f = imgr + ((size_t)v * 3 + c) * HW2D;
        float s = f[o00] * m00 + f[o01] * m01 + f[o10] * m10 + f[o11] * m11;
        vol41[(size_t)(3 * v + c) * NVOL + idx] = s;
    }
    if (inm) count[idx] += 1.0f;
}

// var = sq/cnt - (sum/cnt)^2, in place over vol41 channels 9..40
__global__ void k_var_finalize(float* __restrict__ vol41, const float* __restrict__ volsq,
                               const float* __restrict__ count) {
    int idx = blockIdx.x * blockDim.x + threadIdx.x;
    if (idx >= NVOL) return;
    float inv = 1.0f / count[idx];
    for (int c = 0; c < 32; ++c) {
        float s = vol41[(size_t)(9 + c) * NVOL + idx];
        float q = volsq[(size_t)c * NVOL + idx];
        float sm = s * inv;
        vol41[(size_t)(9 + c) * NVOL + idx] = q * inv - sm * sm;
    }
}

// feats [32][3][HW2D] -> d_out region (1,3,32,64,80)
__global__ void k_feats_out(const float* __restrict__ feats, float* __restrict__ out) {
    int idx = blockIdx.x * blockDim.x + threadIdx.x;
    if (idx >= 3 * 32 * HW2D) return;
    int v = idx / (32 * HW2D);
    int r = idx % (32 * HW2D);
    int c = r / HW2D, p = r % HW2D;
    out[idx] = feats[((size_t)c * 3 + v) * HW2D + p];
}

// ---------------------------------------------------------------------------
// Host-side orchestration
// ---------------------------------------------------------------------------
struct Ctx {
    float* ws; hipStream_t stream; float* stats; _Float16* packW;
};

static void run_conv(Ctx& cx, const float* X, const float* Wraw, const float* bias,
                     float* Y, int M, int Cin, int kd, int kh, int kw,
                     int inD, int inH, int inW, int outD, int outH, int outW,
                     int sd, int sh, int sw, int pd, int ph, int pw, int deconv) {
    int K = Cin * kd * kh * kw;
    int Kp = ceil_div(K, 32) * 32;
    int M16 = ceil_div(M, 16) * 16;
    int tot = M16 * Kp;
    k_pack_weights<<<ceil_div(tot, 256), 256, 0, cx.stream>>>(
        Wraw, cx.packW, M, K, Kp, M16, Cin, kd, kh, kw, deconv);
    int N = outD * outH * outW;
    dim3 grid(ceil_div(N, 64), M16 / 16);
    if (deconv) {
        k_conv_gemm_wmma<1><<<grid, 128, 0, cx.stream>>>(
            X, cx.packW, bias, Y, M, Kp, K, Cin, inD, inH, inW,
            outD, outH, outW, kd, kh, kw, sd, sh, sw, pd, ph, pw);
    } else {
        k_conv_gemm_wmma<0><<<grid, 128, 0, cx.stream>>>(
            X, cx.packW, bias, Y, M, Kp, K, Cin, inD, inH, inW,
            outD, outH, outW, kd, kh, kw, sd, sh, sw, pd, ph, pw);
    }
}

static void run_bn(Ctx& cx, const float* X, const float* g, const float* b,
                   const float* skip, float* Y, int C, int N) {
    hipMemsetAsync(cx.stats, 0, (size_t)2 * C * sizeof(float), cx.stream);
    int gx = (N + 2047) / 2048; if (gx < 1) gx = 1; if (gx > 1024) gx = 1024;
    dim3 gs(gx, C);
    k_bn_stats<<<gs, 256, 0, cx.stream>>>(X, cx.stats, cx.stats + C, N);
    int ga = (N + 255) / 256; if (ga > 2048) ga = 2048;
    dim3 g2(ga, C);
    k_bn_apply<<<g2, 256, 0, cx.stream>>>(X, cx.stats, cx.stats + C, g, b, skip, Y,
                                          N, 1.0f / (float)N);
}

extern "C" void kernel_launch(void* const* d_in, const int* in_sizes, int n_in,
                              void* d_out, int out_size, void* d_ws, size_t ws_size,
                              hipStream_t stream) {
    (void)in_sizes; (void)n_in; (void)out_size; (void)ws_size;
    // Leaf order (setup_inputs insertion order): imgs, proj_mats, near_far,
    // params{feat[8]{w,g,b}, top_w, top_b, reg[7]{w,g,b}, regT[3]{w,g,b}}
    auto in = [&](int i) { return (const float*)d_in[i]; };
    const float* imgs = in(0);
    const float* proj = in(1);
    const float* nf   = in(2);
    auto fW = [&](int l) { return in(3 + 3 * l); };
    auto fG = [&](int l) { return in(4 + 3 * l); };
    auto fB = [&](int l) { return in(5 + 3 * l); };
    const float* topW = in(27);
    const float* topB = in(28);
    auto rW = [&](int j) { return in(29 + 3 * j); };
    auto rG = [&](int j) { return in(30 + 3 * j); };
    auto rB = [&](int j) { return in(31 + 3 * j); };
    auto tW = [&](int j) { return in(50 + 3 * j); };
    auto tG = [&](int j) { return in(51 + 3 * j); };
    auto tB = [&](int j) { return in(52 + 3 * j); };

    // workspace layout (floats, 64-elt aligned slabs)
    float* W = (float*)d_ws;
    size_t o = 0;
    auto alloc = [&](size_t n) { size_t r = o; o += (n + 63) & ~(size_t)63; return r; };
    size_t F_X0   = alloc((size_t)9 * HWIMG);
    size_t F_PING = alloc(2100000);
    size_t F_PONG = alloc(2100000);
    size_t F_FEAT = alloc((size_t)32 * 3 * HW2D);
    size_t F_IMGR = alloc((size_t)9 * HW2D);
    size_t F_DV   = alloc(128);
    size_t F_ST   = alloc(128);
    size_t F_V41  = alloc((size_t)41 * NVOL);
    size_t F_VSQ  = alloc((size_t)32 * NVOL);
    size_t F_CNT  = alloc(NVOL);
    size_t F_G1   = alloc((size_t)8 * NVOL);
    size_t F_C0   = alloc((size_t)8 * NVOL);
    size_t F_B1   = alloc((size_t)16 * 81920);
    size_t F_C2   = alloc((size_t)16 * 81920);
    size_t F_B3   = alloc((size_t)32 * 10240);
    size_t F_C4   = alloc((size_t)32 * 10240);
    size_t F_B5   = alloc((size_t)64 * 1280);
    size_t F_X6   = alloc((size_t)64 * 1280);
    size_t F_U0   = alloc((size_t)32 * 10240);
    size_t F_U1   = alloc((size_t)16 * 81920);
    size_t F_PK   = alloc(131072);   // 262144 f16 packed-weight slots

    Ctx cx{W, stream, W + F_ST, (_Float16*)(W + F_PK)};
    float* X0   = W + F_X0;   float* PING = W + F_PING; float* PONG = W + F_PONG;
    float* FEAT = W + F_FEAT; float* IMGR = W + F_IMGR; float* DV   = W + F_DV;
    float* V41  = W + F_V41;  float* VSQ  = W + F_VSQ;  float* CNT  = W + F_CNT;
    float* G1   = W + F_G1;   float* C0   = W + F_C0;   float* B1   = W + F_B1;
    float* C2   = W + F_C2;   float* B3   = W + F_B3;   float* C4   = W + F_C4;
    float* B5   = W + F_B5;   float* X6   = W + F_X6;   float* U0   = W + F_U0;
    float* U1   = W + F_U1;

    float* dout       = (float*)d_out;
    float* dout_feats = dout + (size_t)8 * NVOL;
    float* dout_depth = dout_feats + (size_t)3 * 32 * HW2D;

    // ---- feature net (views folded into depth axis, kd=1) ------------------
    k_transpose_imgs<<<ceil_div(9 * HWIMG, 256), 256, 0, stream>>>(imgs, X0);

    struct L2D { int co, ci, k, st, pd, iH, iW, oH, oW; };
    const L2D FL[8] = {
        {8, 3, 3, 1, 1, 256, 320, 256, 320},  {8, 8, 3, 1, 1, 256, 320, 256, 320},
        {16, 8, 5, 2, 2, 256, 320, 128, 160}, {16, 16, 3, 1, 1, 128, 160, 128, 160},
        {16, 16, 3, 1, 1, 128, 160, 128, 160},{32, 16, 5, 2, 2, 128, 160, 64, 80},
        {32, 32, 3, 1, 1, 64, 80, 64, 80},    {32, 32, 3, 1, 1, 64, 80, 64, 80}};
    const float* cur = X0;
    for (int l = 0; l < 8; ++l) {
        const L2D& L = FL[l];
        run_conv(cx, cur, fW(l), nullptr, PONG, L.co, L.ci, 1, L.k, L.k,
                 3, L.iH, L.iW, 3, L.oH, L.oW, 1, L.st, L.st, 0, L.pd, L.pd, 0);
        run_bn(cx, PONG, fG(l), fB(l), nullptr, PING, L.co, 3 * L.oH * L.oW);
        cur = PING;
    }
    // top 1x1 conv + bias -> FEAT [32][3][HW2D]
    run_conv(cx, cur, topW, topB, FEAT, 32, 32, 1, 1, 1,
             3, HF, WF, 3, HF, WF, 1, 1, 1, 0, 0, 0, 0);

    // ---- cost volume -------------------------------------------------------
    k_resize_imgs<<<ceil_div(9 * HW2D, 256), 256, 0, stream>>>(imgs, IMGR);
    k_depth_vals<<<1, 128, 0, stream>>>(nf, DV, dout_depth);
    k_init_vol<<<ceil_div(NVOL, 256), 256, 0, stream>>>(FEAT, IMGR, V41, VSQ, CNT);
    for (int v = 1; v < 3; ++v)
        k_warp_view<<<ceil_div(NVOL, 256), 256, 0, stream>>>(FEAT, IMGR, proj, DV,
                                                             V41, VSQ, CNT, v);
    k_var_finalize<<<ceil_div(NVOL, 256), 256, 0, stream>>>(V41, VSQ, CNT);

    // ---- cost regularization (3D U-Net) ------------------------------------
    struct L3D { int co, ci, st, iD, iH, iW, oD, oH, oW; };
    const L3D RL[7] = {
        {8, 41, 1, 128, 64, 80, 128, 64, 80}, {16, 8, 2, 128, 64, 80, 64, 32, 40},
        {16, 16, 1, 64, 32, 40, 64, 32, 40},  {32, 16, 2, 64, 32, 40, 32, 16, 20},
        {32, 32, 1, 32, 16, 20, 32, 16, 20},  {64, 32, 2, 32, 16, 20, 16, 8, 10},
        {64, 64, 1, 16, 8, 10, 16, 8, 10}};
    const float* rin[7]  = {V41, C0, B1, C2, B3, C4, B5};
    float*       rout[7] = {C0, B1, C2, B3, C4, B5, X6};
    for (int j = 0; j < 7; ++j) {
        const L3D& L = RL[j];
        run_conv(cx, rin[j], rW(j), nullptr, G1, L.co, L.ci, 3, 3, 3,
                 L.iD, L.iH, L.iW, L.oD, L.oH, L.oW,
                 L.st, L.st, L.st, 1, 1, 1, 0);
        run_bn(cx, G1, rG(j), rB(j), nullptr, rout[j], L.co,
               L.oD * L.oH * L.oW);
    }
    // decoder: deconv3d + BN + lrelu + skip add
    run_conv(cx, X6, tW(0), nullptr, G1, 32, 64, 3, 3, 3,
             16, 8, 10, 32, 16, 20, 1, 1, 1, 1, 1, 1, 1);
    run_bn(cx, G1, tG(0), tB(0), C4, U0, 32, 32 * 16 * 20);
    run_conv(cx, U0, tW(1), nullptr, G1, 16, 32, 3, 3, 3,
             32, 16, 20, 64, 32, 40, 1, 1, 1, 1, 1, 1, 1);
    run_bn(cx, G1, tG(1), tB(1), C2, U1, 16, 64 * 32 * 40);
    run_conv(cx, U1, tW(2), nullptr, G1, 8, 16, 3, 3, 3,
             64, 32, 40, 128, 64, 80, 1, 1, 1, 1, 1, 1, 1);
    run_bn(cx, G1, tG(2), tB(2), C0, dout, 8, NVOL);   // final vol -> d_out

    // feats_l -> d_out (1,3,32,64,80)
    k_feats_out<<<ceil_div(3 * 32 * HW2D, 256), 256, 0, stream>>>(FEAT, dout_feats);
}
